// VectorQuantizerWithChannel_49134425866706
// MI455X (gfx1250) — matrix operand order repaired and checked by reading, not compiled, not run on hardware
//
#include <hip/hip_runtime.h>
#include <stdint.h>

typedef __attribute__((ext_vector_type(16))) __bf16         v16bf;
typedef __attribute__((ext_vector_type(16))) unsigned short v16u;
typedef __attribute__((ext_vector_type(8)))  float          v8f;

#define N_E      1024
#define E_DIM    128
#define NUM_BITS 10
#define BATCH    4096
#define HW       25
#define M_TOTAL  (BATCH * HW)   /* 102400 rows */
#define NOISE_STD 0.22360679774997896f  /* sqrt(0.5 / 10^(10/10)) */
#define LOSS_SCALE (1.25f / 13107200.0f) /* (1+BETA) / (B*C*H*W) */

__device__ __forceinline__ unsigned short f32_to_bf16(float f) {
  uint32_t u = __float_as_uint(f);
  u += 0x7FFFu + ((u >> 16) & 1u);      // round-to-nearest-even
  return (unsigned short)(u >> 16);
}
__device__ __forceinline__ float bf16_to_f32(unsigned short h) {
  return __uint_as_float(((uint32_t)h) << 16);
}

// ---------------------------------------------------------------------------
// Prep: split codebook into bf16 hi/lo, compute ||e||^2, zero the loss slot.
// 1024 blocks x 128 threads (one code per block, one K element per thread).
// ---------------------------------------------------------------------------
__global__ __launch_bounds__(E_DIM) void vq_prep(
    const float* __restrict__ emb,
    unsigned short* __restrict__ ehi,
    unsigned short* __restrict__ elo,
    float* __restrict__ enorm,
    float* __restrict__ loss_slot) {
  const int n = blockIdx.x;
  const int k = threadIdx.x;
  float w = emb[n * E_DIM + k];
  unsigned short hb = f32_to_bf16(w);
  float hf = bf16_to_f32(hb);
  ehi[n * E_DIM + k] = hb;
  elo[n * E_DIM + k] = f32_to_bf16(w - hf);

  __shared__ float red[E_DIM];
  red[k] = w * w;
  __syncthreads();
  #pragma unroll
  for (int s = E_DIM / 2; s > 0; s >>= 1) {
    if (k < s) red[k] += red[k + s];
    __syncthreads();
  }
  if (k == 0) enorm[n] = red[0];
  if (n == 0 && k == 0) loss_slot[0] = 0.0f;  // re-zero every call (graph-safe)
}

// ---------------------------------------------------------------------------
// Main: 800 blocks x 256 threads (8 waves). Each wave owns 16 z-rows, sweeps
// the full 1024-code book in 16-wide N tiles with split-bf16 WMMA, does a
// wave-local argmin, then channel sim + output + loss epilogue.
// ---------------------------------------------------------------------------
__global__ __launch_bounds__(256) void vq_main(
    const float* __restrict__ z,
    const float* __restrict__ emb,
    const float* __restrict__ noise,
    const unsigned short* __restrict__ ehi,
    const unsigned short* __restrict__ elo,
    const float* __restrict__ enorm,
    float* __restrict__ out) {   // out[0] = loss, out+1 = [B,128,5,5]
  const int wave = threadIdx.x >> 5;
  const int lane = threadIdx.x & 31;
  const int half = lane >> 4;
  const int l15  = lane & 15;
  const int m0   = blockIdx.x * 128 + wave * 16;  // this wave's 16 rows

  // ---- Build A fragments (16 rows x K=128, bf16 hi + lo) ------------------
  // A 16-bit layout (16x32/frag): lane m = l15; K = kb + 16*half + j
  // (16 contiguous K per half, matching B).
  v16u ah[4], al[4];
  {
    const int mr = m0 + l15;
    const int b  = mr / HW, hw = mr % HW;
    const float* zrow = z + (size_t)b * E_DIM * HW + hw;  // elem k at [k*HW]
    #pragma unroll
    for (int kc = 0; kc < 4; ++kc) {
      const int kb = kc * 32;
      v16u h, l;
      #pragma unroll
      for (int j = 0; j < 16; ++j) {
        const int k = kb + 16 * half + j;
        float w = zrow[(size_t)k * HW];
        unsigned short hb = f32_to_bf16(w);
        h[j] = hb;
        l[j] = f32_to_bf16(w - bf16_to_f32(hb));
      }
      ah[kc] = h;
      al[kc] = l;
    }
  }

  // ---- N sweep: score = ||e||^2 - 2 z.e via 3-product split-bf16 WMMA -----
  float best[8];
  int   bestn[8];
  #pragma unroll
  for (int v = 0; v < 8; ++v) { best[v] = 3.4e38f; bestn[v] = 0; }

  // Unroll by 2: two independent tiles in flight so one tile's B loads issue
  // under the other tile's WMMA chain (cuts per-WMMA s_wait_loadcnt stalls).
  #pragma unroll 2
  for (int nb = 0; nb < N_E; nb += 16) {
    const int nrow = nb + l15;  // B lane holds column n = lane%16
    const unsigned short* eh = ehi + (size_t)nrow * E_DIM;
    const unsigned short* el = elo + (size_t)nrow * E_DIM;
    // Branch-free next-tile warm-up (wraps to tile 0 at the end; harmless).
    {
      const size_t pn = (size_t)(((nb + 16) & (N_E - 1)) + l15) * E_DIM;
      __builtin_prefetch(ehi + pn, 0, 1);   // global_prefetch_b8
      __builtin_prefetch(elo + pn, 0, 1);
    }
    v8f c0 = {}, c1 = {};  // two chains to halve WMMA RAW depth
    #pragma unroll
    for (int kc = 0; kc < 4; ++kc) {
      // B 16-bit layout (32x16): lane n = l15, 16 contiguous K per half.
      const int ko = kc * 32 + 16 * half;
      v16u bhv = *(const v16u*)(eh + ko);
      v16u blv = *(const v16u*)(el + ko);
      v16bf Ah = __builtin_bit_cast(v16bf, ah[kc]);
      v16bf Al = __builtin_bit_cast(v16bf, al[kc]);
      v16bf Bh = __builtin_bit_cast(v16bf, bhv);
      v16bf Bl = __builtin_bit_cast(v16bf, blv);
      c0 = __builtin_amdgcn_wmma_f32_16x16x32_bf16(false, Ah, false, Bh,
                                                   (short)0, c0, false, false);
      c1 = __builtin_amdgcn_wmma_f32_16x16x32_bf16(false, Al, false, Bh,
                                                   (short)0, c1, false, false);
      c1 = __builtin_amdgcn_wmma_f32_16x16x32_bf16(false, Ah, false, Bl,
                                                   (short)0, c1, false, false);
    }
    const float en = enorm[nrow];
    #pragma unroll
    for (int v = 0; v < 8; ++v) {  // C layout: lane=N, VGPR v = M (+8 if half1)
      float s = fmaf(-2.0f, c0[v] + c1[v], en);
      if (s < best[v]) { best[v] = s; bestn[v] = nrow; }
    }
  }

  // ---- argmin across the 16 N-lanes of each half (xor<16 keeps halves) ----
  #pragma unroll
  for (int d = 1; d < 16; d <<= 1) {
    #pragma unroll
    for (int v = 0; v < 8; ++v) {
      float os = __shfl_xor(best[v], d, 32);
      int   oi = __shfl_xor(bestn[v], d, 32);
      if (os < best[v] || (os == best[v] && oi < bestn[v])) {
        best[v] = os; bestn[v] = oi;
      }
    }
  }

  __shared__ int s_idx[8][16];
  __shared__ int s_ridx[8][16];
  if (l15 == 0) {  // lane0 -> rows 0..7, lane16 -> rows 8..15
    #pragma unroll
    for (int v = 0; v < 8; ++v) s_idx[wave][half * 8 + v] = bestn[v];
  }
  __syncthreads();

  // ---- channel simulation: one row per lane (lanes 0..15) -----------------
  if (lane < 16) {
    const int m = m0 + lane;
    const int idx = s_idx[wave][lane];
    const float* nz = noise + (size_t)m * NUM_BITS;
    int rec = 0;
    #pragma unroll
    for (int j = 0; j < NUM_BITS; ++j) {
      const int sh = NUM_BITS - 1 - j;           // MSB first
      float bip = ((idx >> sh) & 1) ? 1.0f : -1.0f;
      float r = bip + nz[j] * NOISE_STD;
      if (r >= 0.0f) rec |= (1 << sh);
    }
    s_ridx[wave][lane] = rec;
  }
  __syncthreads();

  // ---- epilogue: out = E[r_idx] (NCHW), loss += ||E[idx]-z||^2 ------------
  float acc = 0.0f;
  for (int r = 0; r < 16; ++r) {
    const int m  = m0 + r;
    const int b  = m / HW, hw = m % HW;
    const float* eq = emb + (size_t)s_idx[wave][r] * E_DIM;
    const float* er = emb + (size_t)s_ridx[wave][r] * E_DIM;
    const float* zr = z + (size_t)b * E_DIM * HW + hw;
    float* orow = out + 1 + (size_t)b * E_DIM * HW + hw;
    #pragma unroll
    for (int ci = 0; ci < 4; ++ci) {
      const int c = lane + 32 * ci;
      float d = eq[c] - zr[(size_t)c * HW];
      acc += d * d;
      orow[(size_t)c * HW] = er[c];
    }
  }
  #pragma unroll
  for (int d = 16; d >= 1; d >>= 1) acc += __shfl_xor(acc, d, 32);
  if (lane == 0) atomicAdd(out, acc * LOSS_SCALE);
}

// ---------------------------------------------------------------------------
extern "C" void kernel_launch(void* const* d_in, const int* in_sizes, int n_in,
                              void* d_out, int out_size, void* d_ws,
                              size_t ws_size, hipStream_t stream) {
  const float* z     = (const float*)d_in[0];
  const float* emb   = (const float*)d_in[1];
  const float* noise = (const float*)d_in[2];
  float* out = (float*)d_out;

  unsigned short* ehi = (unsigned short*)d_ws;             // 256 KB
  unsigned short* elo = ehi + (size_t)N_E * E_DIM;         // 256 KB
  float* enorm = (float*)(elo + (size_t)N_E * E_DIM);      // 4 KB

  vq_prep<<<N_E, E_DIM, 0, stream>>>(emb, ehi, elo, enorm, out);
  vq_main<<<M_TOTAL / 128, 256, 0, stream>>>(z, emb, noise, ehi, elo, enorm,
                                             out);
}
